// SpaceAttnBlock_7430293422359
// MI455X (gfx1250) — matrix operand order repaired and checked by reading, not compiled, not run on hardware
//
#include <hip/hip_runtime.h>
#include <hip/hip_bf16.h>

// ---------------------------------------------------------------------------
// SpaceAttnBlock for MI455X (gfx1250, wave32, WMMA).
// B=8, C=256, H=64, W=128  ->  L=8192, rows M = B*L = 65536, nhead=8, dh=32.
// GEMMs: bf16 v_wmma_f32_16x16x32_bf16, A tiles staged via CDNA5 async
// global->LDS copies (ASYNCcnt) with double buffering.
// ---------------------------------------------------------------------------

typedef __bf16 bf16;
typedef __attribute__((ext_vector_type(8)))  bf16  v8bf;
typedef __attribute__((ext_vector_type(16))) bf16  v16bf;
typedef __attribute__((ext_vector_type(8)))  float v8f;
typedef __attribute__((ext_vector_type(4)))  int   v4i;

#define C_DIM   256
#define L_DIM   8192
#define B_DIM   8
#define NHEAD   8
#define DH      32
#define M_ROWS  (B_DIM * L_DIM)   // 65536

#define BM 128
#define BN 64
#define BK 32

#define GLOBAL_AS __attribute__((address_space(1)))
#define LDS_AS    __attribute__((address_space(3)))

#ifndef __has_builtin
#define __has_builtin(x) 0
#endif

enum { EP_F32 = 0, EP_ELU1 = 1, EP_SCALE = 2, EP_RELU = 3 };

// ---- CDNA5 async global->LDS copy (16B per lane), tracked by ASYNCcnt ----
__device__ __forceinline__ void async_copy_b128(const bf16* gsrc, bf16* ldst) {
#if __has_builtin(__builtin_amdgcn_global_load_async_to_lds_b128)
    __builtin_amdgcn_global_load_async_to_lds_b128(
        (GLOBAL_AS v4i*)(uintptr_t)gsrc,
        (LDS_AS v4i*)(unsigned)(uintptr_t)ldst, 0, 0);
#else
    unsigned lds_off = (unsigned)(uintptr_t)ldst;
    unsigned long long ga = (unsigned long long)(uintptr_t)gsrc;
    asm volatile("global_load_async_to_lds_b128 %0, %1, off"
                 :: "v"(lds_off), "v"(ga) : "memory");
#endif
}

__device__ __forceinline__ void async_wait0() {
#if __has_builtin(__builtin_amdgcn_s_wait_asynccnt)
    __builtin_amdgcn_s_wait_asynccnt(0);
#else
    asm volatile("s_wait_asynccnt 0" ::: "memory");
#endif
}

__device__ __forceinline__ v16bf cat8(v8bf lo, v8bf hi) {
    v16bf r;
#pragma unroll
    for (int e = 0; e < 8; ++e) { r[e] = lo[e]; r[8 + e] = hi[e]; }
    return r;
}

// ---------------------------------------------------------------------------
// [B,C,L] f32  ->  [B*L, C] as f32 (residual) and bf16 (GEMM operand)
// ---------------------------------------------------------------------------
__global__ __launch_bounds__(256)
void k_transpose_in(const float* __restrict__ x,
                    float* __restrict__ xsf, bf16* __restrict__ xsb) {
    __shared__ float tile[32][33];
    const int b  = blockIdx.z;
    const int c0 = blockIdx.y * 32;
    const int l0 = blockIdx.x * 32;
    const int tx = threadIdx.x & 31;
    const int ty = threadIdx.x >> 5;   // 0..7
    const float* src = x + ((size_t)b * C_DIM + c0) * L_DIM + l0;
#pragma unroll
    for (int i = 0; i < 4; ++i) {
        int c = ty + i * 8;
        tile[c][tx] = src[(size_t)c * L_DIM + tx];
    }
    __syncthreads();
#pragma unroll
    for (int i = 0; i < 4; ++i) {
        int l = ty + i * 8;
        float v = tile[tx][l];
        size_t o = ((size_t)b * L_DIM + l0 + l) * C_DIM + c0 + tx;
        xsf[o] = v;
        xsb[o] = (bf16)v;
    }
}

// ---------------------------------------------------------------------------
// Weight prep: f32 [K,N] row-major -> bf16 [N,K] (transposed): each WMMA
// B-fragment becomes two contiguous b128 loads per lane.
// ---------------------------------------------------------------------------
__global__ __launch_bounds__(256)
void k_wprep(const float* __restrict__ src, bf16* __restrict__ dst, int K, int N) {
    int i = blockIdx.x * 256 + threadIdx.x;
    if (i < K * N) {
        int k = i / N, n = i % N;
        dst[n * K + k] = (bf16)src[i];
    }
}

__global__ __launch_bounds__(256)
void k_zero(float* __restrict__ p, int n) {
    int i = blockIdx.x * 256 + threadIdx.x;
    if (i < n) p[i] = 0.0f;
}

// ---------------------------------------------------------------------------
// bf16 WMMA GEMM:  C[M,N] = A[M,K] @ W[K,N], W pre-transposed as WT[N,K].
// Block tile 128x64, BK=32, 8 waves in 4(M) x 2(N), wave tile 32x32 (2x2 WMMA).
// A tiles: async global->LDS, double buffered, one barrier per K-step.
// B fragments: direct from cache-resident WT.
// SPLITA: A = [A0 | A1] along K at k=256 (fused concat, both lda=256).
// ---------------------------------------------------------------------------
template <int EP, bool SPLITA>
__global__ __launch_bounds__(256)
void k_gemm_bf16_wmma(const bf16* __restrict__ A0, const bf16* __restrict__ A1,
                      const bf16* __restrict__ WT,
                      float* __restrict__ Cf, bf16* __restrict__ Cb,
                      int M, int N, int K, float scale) {
    __shared__ bf16 ldsA[2][BM * BK];  // 2 x 8 KB

    const int t    = threadIdx.x;
    const int wave = t >> 5;
    const int lane = t & 31;
    const int wm   = wave & 3;         // wave position in M (0..3)
    const int wn   = wave >> 2;        // wave position in N (0..1)
    const int bm   = blockIdx.y * BM;
    const int bn   = blockIdx.x * BN;
    const int lm   = lane & 15;
    const int lh   = lane >> 4;

    // staging coordinates: 256 threads x 2 chunks of 8 bf16 cover 128x32
    const int srow = t >> 2;           // 0..63
    const int scol = (t & 3) << 3;     // 0,8,16,24

    auto stage = [&](int k0, int buf) {
#pragma unroll
        for (int it = 0; it < 2; ++it) {
            int row = srow + it * 64;
            const bf16* src;
            if (SPLITA) {
                int kg = k0 + scol;
                src = (kg < 256) ? (A0 + (bm + row) * 256 + kg)
                                 : (A1 + (bm + row) * 256 + (kg - 256));
            } else {
                src = A0 + (bm + row) * K + k0 + scol;
            }
            async_copy_b128(src, &ldsA[buf][row * BK + scol]);
        }
    };

    v8f acc[2][2];
#pragma unroll
    for (int i = 0; i < 2; ++i)
#pragma unroll
        for (int j = 0; j < 2; ++j)
#pragma unroll
            for (int p = 0; p < 8; ++p) acc[i][j][p] = 0.0f;

    const int ksteps = K / BK;
    stage(0, 0);
    for (int kk = 0; kk < ksteps; ++kk) {
        const int buf = kk & 1;
        const int k0  = kk * BK;
        async_wait0();                 // my async fills of ldsA[buf] landed
        __syncthreads();               // everyone's landed; prev reads done
        if (kk + 1 < ksteps) stage(k0 + BK, buf ^ 1);  // overlap with compute

        // ---- B fragments straight from global WT[N,K] (contiguous k = 16*lh + s per lane) ----
        v16bf bfrag[2];
#pragma unroll
        for (int j = 0; j < 2; ++j) {
            int gn = bn + wn * 32 + j * 16 + lm;
            const bf16* wr = WT + gn * K + k0 + lh * 16;
            bfrag[j] = cat8(*(const v8bf*)wr, *(const v8bf*)(wr + 8));
        }
        // ---- A fragments from LDS (contiguous k = 16*lh + s per lane) ----
        v16bf afrag[2];
#pragma unroll
        for (int i = 0; i < 2; ++i) {
            int row = wm * 32 + i * 16 + lm;
            v8bf lo = *(const v8bf*)&ldsA[buf][row * BK + lh * 16];
            v8bf hi = *(const v8bf*)&ldsA[buf][row * BK + lh * 16 + 8];
            afrag[i] = cat8(lo, hi);
        }
        // ---- 4 x v_wmma_f32_16x16x32_bf16 ----
#pragma unroll
        for (int i = 0; i < 2; ++i)
#pragma unroll
            for (int j = 0; j < 2; ++j)
                acc[i][j] = __builtin_amdgcn_wmma_f32_16x16x32_bf16(
                    false, afrag[i], false, bfrag[j],
                    (short)0, acc[i][j], false, false);
    }

    // ---- epilogue (C/D 16x16 f32 layout: N = lane&15, M = (lane>>4)*8+p) ----
#pragma unroll
    for (int i = 0; i < 2; ++i)
#pragma unroll
        for (int j = 0; j < 2; ++j)
#pragma unroll
            for (int p = 0; p < 8; ++p) {
                int gm = bm + wm * 32 + i * 16 + lh * 8 + p;
                int gn = bn + wn * 32 + j * 16 + lm;
                float xv = acc[i][j][p];
                int o = gm * N + gn;              // < 2^31, 32-bit math
                if (EP == EP_F32) {
                    Cf[o] = xv;
                } else if (EP == EP_ELU1) {       // elu(x)+1
                    Cb[o] = (bf16)((xv > 0.0f) ? (xv + 1.0f) : __expf(xv));
                } else if (EP == EP_SCALE) {      // v / L
                    Cb[o] = (bf16)(xv * scale);
                } else {                          // relu
                    Cb[o] = (bf16)fmaxf(xv, 0.0f);
                }
            }
}

// ---------------------------------------------------------------------------
// KV[b,h] = sum_l K[l]^T outer (V/L)[l]  (32x32 f32) and Ksum[b,h] (32 f32).
// grid = (B*NHEAD, 16 L-splits); f32 atomics combine splits.
// ---------------------------------------------------------------------------
__global__ __launch_bounds__(256)
void k_kv_reduce(const bf16* __restrict__ Kf, const bf16* __restrict__ Vf,
                 float* __restrict__ KV, float* __restrict__ Ksum) {
    __shared__ bf16 sK[64 * 32], sV[64 * 32];
    const int bh    = blockIdx.x;            // 0..63
    const int b     = bh >> 3, h = bh & 7;
    const int split = blockIdx.y;            // 0..15
    const int t     = threadIdx.x;
    const int dv    = t & 31;
    const int g     = t >> 5;                // 0..7 -> 4 dk each
    const size_t base = (size_t)b * L_DIM * C_DIM + h * DH;
    const int l0 = split * 512;

    float acc[4] = {0.f, 0.f, 0.f, 0.f};
    float ks = 0.f;

    const int crow = t >> 2;                 // 0..63
    const int ccol = (t & 3) << 3;           // 0,8,16,24
    for (int s = 0; s < 512; s += 64) {
        size_t gidx = base + (size_t)(l0 + s + crow) * C_DIM + ccol;
        *(v8bf*)&sK[crow * 32 + ccol] = *(const v8bf*)(Kf + gidx);
        *(v8bf*)&sV[crow * 32 + ccol] = *(const v8bf*)(Vf + gidx);
        __syncthreads();
#pragma unroll 4
        for (int l = 0; l < 64; ++l) {
            float vv = (float)sV[l * 32 + dv];
#pragma unroll
            for (int j = 0; j < 4; ++j)
                acc[j] += (float)sK[l * 32 + g * 4 + j] * vv;
        }
        if (t < 32) {
#pragma unroll 4
            for (int l = 0; l < 64; ++l) ks += (float)sK[l * 32 + t];
        }
        __syncthreads();
    }
    float* kvb = KV + bh * DH * DH;
#pragma unroll
    for (int j = 0; j < 4; ++j)
        atomicAdd(&kvb[(g * 4 + j) * DH + dv], acc[j]);
    if (t < 32) atomicAdd(&Ksum[bh * DH + t], ks);
}

// ---------------------------------------------------------------------------
// msg[l,h,:] = (Q[l,h,:] @ KV[b,h]) * L / (Q.Ksum + eps).  Wave per head.
// ---------------------------------------------------------------------------
__global__ __launch_bounds__(256)
void k_attn_apply(const bf16* __restrict__ Qf, const float* __restrict__ KV,
                  const float* __restrict__ Ksum, bf16* __restrict__ out) {
    const int r    = blockIdx.x;             // 0..65535
    const int b    = r >> 13;                // /8192
    const int head = threadIdx.x >> 5;
    const int dv   = threadIdx.x & 31;
    const int bh   = b * NHEAD + head;
    const bf16*  q  = Qf + r * C_DIM + head * DH;
    const float* kv = KV + bh * DH * DH;
    const float* ks = Ksum + bh * DH;

    // pull the whole 32-element Q row as 4 x b128 into registers
    v8bf qv[4];
#pragma unroll
    for (int c = 0; c < 4; ++c) qv[c] = *(const v8bf*)(q + c * 8);

    float part = (float)q[dv] * ks[dv];
#pragma unroll
    for (int off = 16; off > 0; off >>= 1) part += __shfl_xor(part, off, 32);
    float z = (float)L_DIM / (part + 1e-6f);

    float acc = 0.0f;
#pragma unroll
    for (int c = 0; c < 4; ++c)
#pragma unroll
        for (int e = 0; e < 8; ++e)
            acc += (float)qv[c][e] * kv[(c * 8 + e) * DH + dv];

    out[r * C_DIM + head * DH + dv] = (bf16)(acc * z);
}

// ---------------------------------------------------------------------------
// Row LayerNorm over C=256.  MODE 0: -> bf16.  MODE 1: residual + output
// transpose back to [B,C,H,W] f32.
// ---------------------------------------------------------------------------
template <int MODE>
__global__ __launch_bounds__(256)
void k_ln_row(const float* __restrict__ in, const float* __restrict__ g,
              const float* __restrict__ bb, bf16* __restrict__ outb,
              const float* __restrict__ resid, float* __restrict__ outf) {
    __shared__ float s1[256], s2[256];
    const int r = blockIdx.x;
    const int c = threadIdx.x;
    float v = in[(size_t)r * C_DIM + c];
    s1[c] = v; s2[c] = v * v;
    __syncthreads();
#pragma unroll
    for (int off = 128; off > 0; off >>= 1) {
        if (c < off) { s1[c] += s1[c + off]; s2[c] += s2[c + off]; }
        __syncthreads();
    }
    float mean = s1[0] * (1.0f / 256.0f);
    float var  = s2[0] * (1.0f / 256.0f) - mean * mean;
    float y = (v - mean) * rsqrtf(var + 1e-5f) * g[c] + bb[c];
    if (MODE == 0) {
        outb[(size_t)r * C_DIM + c] = (bf16)y;
    } else {
        float res = resid[(size_t)r * C_DIM + c] + y;
        int b = r >> 13, l = r & (L_DIM - 1);
        outf[((size_t)b * C_DIM + c) * L_DIM + l] = res;
    }
}

// ---------------------------------------------------------------------------
extern "C" void kernel_launch(void* const* d_in, const int* in_sizes, int n_in,
                              void* d_out, int out_size, void* d_ws, size_t ws_size,
                              hipStream_t stream) {
    const float* x    = (const float*)d_in[0];
    const float* Wq   = (const float*)d_in[1];
    const float* Wk   = (const float*)d_in[2];
    const float* Wv   = (const float*)d_in[3];
    const float* Wm   = (const float*)d_in[4];
    const float* ln1g = (const float*)d_in[5];
    const float* ln1b = (const float*)d_in[6];
    const float* W1   = (const float*)d_in[7];
    const float* W2   = (const float*)d_in[8];
    const float* ln2g = (const float*)d_in[9];
    const float* ln2b = (const float*)d_in[10];
    (void)in_sizes; (void)n_in; (void)out_size; (void)ws_size;

    // workspace layout (256B aligned bump allocator with explicit aliasing)
    char*  w   = (char*)d_ws;
    size_t off = 0;
    auto alloc = [&](size_t bytes) -> void* {
        void* p = w + off;
        off = (off + bytes + 255) & ~(size_t)255;
        return p;
    };
    bf16* WqT = (bf16*)alloc((size_t)256 * 256 * 2);
    bf16* WkT = (bf16*)alloc((size_t)256 * 256 * 2);
    bf16* WvT = (bf16*)alloc((size_t)256 * 256 * 2);
    bf16* WmT = (bf16*)alloc((size_t)256 * 256 * 2);
    bf16* W1T = (bf16*)alloc((size_t)512 * 512 * 2);
    bf16* W2T = (bf16*)alloc((size_t)512 * 256 * 2);
    bf16*  xsb = (bf16*) alloc((size_t)M_ROWS * C_DIM * 2);   // 32 MB
    float* xsf = (float*)alloc((size_t)M_ROWS * C_DIM * 4);   // 64 MB
    bf16*  Qf  = (bf16*) alloc((size_t)M_ROWS * C_DIM * 2);   // 32 MB
    bf16*  Kf  = (bf16*) alloc((size_t)M_ROWS * C_DIM * 2);   // 32 MB
    bf16*  Vf  = (bf16*) alloc((size_t)M_ROWS * C_DIM * 2);   // 32 MB
    float* KV  = (float*)alloc((size_t)64 * DH * DH * 4);     // 256 KB
    float* Ks  = (float*)alloc((size_t)64 * DH * 4);          // 8 KB
    bf16*  att = (bf16*) alloc((size_t)M_ROWS * C_DIM * 2);   // 32 MB
    bf16*  h1  = (bf16*) alloc((size_t)M_ROWS * 512 * 2);     // 64 MB
    // aliases (lifetimes do not overlap):
    float* g1   = (float*)Qf;   // 64 MB over Qf+Kf (after attention done)
    bf16*  msg1 = (bf16*)Vf;    // 32 MB over Vf  (after KV reduction done)
    float* h2   = g1;           // 64 MB reuse (after LN1 consumed g1)

    // 1) weight prep (transpose + bf16 cast)
    k_wprep<<<(256 * 256 + 255) / 256, 256, 0, stream>>>(Wq, WqT, 256, 256);
    k_wprep<<<(256 * 256 + 255) / 256, 256, 0, stream>>>(Wk, WkT, 256, 256);
    k_wprep<<<(256 * 256 + 255) / 256, 256, 0, stream>>>(Wv, WvT, 256, 256);
    k_wprep<<<(256 * 256 + 255) / 256, 256, 0, stream>>>(Wm, WmT, 256, 256);
    k_wprep<<<(512 * 512 + 255) / 256, 256, 0, stream>>>(W1, W1T, 512, 512);
    k_wprep<<<(512 * 256 + 255) / 256, 256, 0, stream>>>(W2, W2T, 512, 256);

    // 2) input transpose [B,C,L] -> [B*L,C] (f32 + bf16)
    k_transpose_in<<<dim3(L_DIM / 32, C_DIM / 32, B_DIM), 256, 0, stream>>>(x, xsf, xsb);

    // 3) zero KV + Ksum (contiguous)
    k_zero<<<(64 * DH * DH + 64 * DH + 255) / 256, 256, 0, stream>>>(KV, 64 * DH * DH + 64 * DH);

    // 4) QKV projections with fused feature maps
    k_gemm_bf16_wmma<EP_ELU1, false><<<dim3(C_DIM / BN, M_ROWS / BM), 256, 0, stream>>>(
        xsb, nullptr, WqT, nullptr, Qf, M_ROWS, C_DIM, C_DIM, 0.f);
    k_gemm_bf16_wmma<EP_ELU1, false><<<dim3(C_DIM / BN, M_ROWS / BM), 256, 0, stream>>>(
        xsb, nullptr, WkT, nullptr, Kf, M_ROWS, C_DIM, C_DIM, 0.f);
    k_gemm_bf16_wmma<EP_SCALE, false><<<dim3(C_DIM / BN, M_ROWS / BM), 256, 0, stream>>>(
        xsb, nullptr, WvT, nullptr, Vf, M_ROWS, C_DIM, C_DIM, 1.0f / (float)L_DIM);

    // 5) linear-attention global reductions + per-token apply
    k_kv_reduce<<<dim3(B_DIM * NHEAD, 16), 256, 0, stream>>>(Kf, Vf, KV, Ks);
    k_attn_apply<<<M_ROWS, 256, 0, stream>>>(Qf, KV, Ks, att);

    // 6) message projection + LN1
    k_gemm_bf16_wmma<EP_F32, false><<<dim3(C_DIM / BN, M_ROWS / BM), 256, 0, stream>>>(
        att, nullptr, WmT, g1, nullptr, M_ROWS, C_DIM, C_DIM, 0.f);
    k_ln_row<0><<<M_ROWS, 256, 0, stream>>>(g1, ln1g, ln1b, msg1, nullptr, nullptr);

    // 7) FFN: relu(concat(xs,msg) @ W1) @ W2, LN2, residual, output transpose
    k_gemm_bf16_wmma<EP_RELU, true><<<dim3(512 / BN, M_ROWS / BM), 256, 0, stream>>>(
        xsb, msg1, W1T, nullptr, h1, M_ROWS, 512, 512, 0.f);
    k_gemm_bf16_wmma<EP_F32, false><<<dim3(C_DIM / BN, M_ROWS / BM), 256, 0, stream>>>(
        h1, nullptr, W2T, h2, nullptr, M_ROWS, C_DIM, 512, 0.f);
    k_ln_row<1><<<M_ROWS, 256, 0, stream>>>(h2, ln2g, ln2b, nullptr, xsf, (float*)d_out);
}
